// MultiheadAttention_83056077570686
// MI455X (gfx1250) — matrix operand order repaired and checked by reading, not compiled, not run on hardware
//
#include <hip/hip_runtime.h>
#include <hip/hip_bf16.h>

typedef __attribute__((ext_vector_type(16))) _Float16 v16h;
typedef __attribute__((ext_vector_type(4)))  _Float16 v4h;
typedef __attribute__((ext_vector_type(8)))  float    v8f;

#define E_DIM 256
#define NHEAD 8
#define HDIM  32
#define BSZ   128
#define TGT   576
#define SRC   256
#define QW    24
#define SIDE  31
#define TDIM  39   // q_h + kv_ws - 1 = 24 + 16 - 1
#define LDA   40   // f16 LDS row stride (80 B: 16-byte multiple, bank-skewed)

// ---------------- CDNA5 async global->LDS helpers ----------------
// GLOBAL_LOAD_ASYNC_TO_LDS_B128: per-lane LDS dest offset in a VGPR, 64-bit
// global address in a VGPR pair; tracked by ASYNCcnt (ISA ch.10 / 15.18.3).

__device__ __forceinline__ void async_copy_b128(void* lds_dst, const void* gsrc) {
  unsigned lds_off = (unsigned)(unsigned long long)lds_dst;  // addr[31:0] = LDS offset
  unsigned long long gaddr = (unsigned long long)gsrc;
  asm volatile("global_load_async_to_lds_b128 %0, %1, off"
               :: "v"(lds_off), "v"(gaddr) : "memory");
}

__device__ __forceinline__ void async_wait0() {
  asm volatile("s_wait_asynccnt 0x0" ::: "memory");
}

// ---------------- WMMA helpers (CDNA5, wave32) ----------------

__device__ __forceinline__ v8f wmma16(v16h a, v16h b, v8f c) {
  // D = A(16x32 f16) * B(32x16 f16) + C(16x16 f32)
  return __builtin_amdgcn_wmma_f32_16x16x32_f16(false, a, false, b, (short)0, c,
                                                false, false);
}

// A fragment 16x32 (MxK) gathered from row-major [row][k] f16 LDS buffer.
// Layout (ISA 7.12.2): M = lane&15; per-lane half h:
//   v = h>>1, slot = h&1, K = 16*(v>>2) + 8*(lane>>4) + 2*(v&3) + slot
__device__ __forceinline__ v16h frag_a(const _Float16* s, int r0, int k0, int ld) {
  const int lane = threadIdx.x & 31;
  const int g = lane >> 4, m = lane & 15;
  v16h a;
#pragma unroll
  for (int h = 0; h < 16; ++h) {
    const int v = h >> 1, sl = h & 1;
    const int k = ((v >> 2) << 4) + (g << 3) + ((v & 3) << 1) + sl;
    a[h] = s[(r0 + m) * ld + k0 + k];
  }
  return a;
}

// Same as frag_a but source is an f32 LDS buffer (attention probabilities).
__device__ __forceinline__ v16h frag_a_f32(const float* s, int r0, int k0, int ld) {
  const int lane = threadIdx.x & 31;
  const int g = lane >> 4, m = lane & 15;
  v16h a;
#pragma unroll
  for (int h = 0; h < 16; ++h) {
    const int v = h >> 1, sl = h & 1;
    const int k = ((v >> 2) << 4) + (g << 3) + ((v & 3) << 1) + sl;
    a[h] = (_Float16)s[(r0 + m) * ld + k0 + k];
  }
  return a;
}

// B fragment 32x16 (KxN) from a [n][k] row-major buffer (B = buf^T).
// Layout: N = lane&15; K = 16*(v>>2) + 8*(lane>>4) + 2*(v&3) + slot
__device__ __forceinline__ v16h frag_bt(const _Float16* s, int n0, int k0, int ld) {
  const int lane = threadIdx.x & 31;
  const int g = lane >> 4, n = lane & 15;
  v16h b;
#pragma unroll
  for (int h = 0; h < 16; ++h) {
    const int v = h >> 1, sl = h & 1;
    const int k = ((v >> 2) << 4) + (g << 3) + ((v & 3) << 1) + sl;
    b[h] = s[(n0 + n) * ld + k0 + k];
  }
  return b;
}

// B fragment 32x16 (KxN) from a [k][n] row-major buffer.
__device__ __forceinline__ v16h frag_bn(const _Float16* s, int k0, int n0, int ld) {
  const int lane = threadIdx.x & 31;
  const int g = lane >> 4, n = lane & 15;
  v16h b;
#pragma unroll
  for (int h = 0; h < 16; ++h) {
    const int v = h >> 1, sl = h & 1;
    const int k = ((v >> 2) << 4) + (g << 3) + ((v & 3) << 1) + sl;
    b[h] = s[(k0 + k) * ld + n0 + n];
  }
  return b;
}

// ---------------- Kernel 1: bicubic resize of RPE table ----------------

__device__ __forceinline__ float cubw(float x) {  // Keys cubic, a = -0.5
  x = fabsf(x);
  if (x <= 1.f) return ((1.5f * x - 2.5f) * x) * x + 1.f;
  if (x < 2.f)  return (((-0.5f * x) + 2.5f) * x - 4.f) * x + 2.f;
  return 0.f;
}

__global__ void rpe_resize_kernel(const float* __restrict__ table,
                                  float* __restrict__ out) {
  const int hh = blockIdx.x;  // head
  const float sc = (float)SIDE / (float)TDIM;
  for (int p = threadIdx.x; p < TDIM * TDIM; p += blockDim.x) {
    const int iy = p / TDIM, ix = p % TDIM;
    const float fy = (iy + 0.5f) * sc - 0.5f;
    const float fx = (ix + 0.5f) * sc - 0.5f;
    const int y0 = (int)floorf(fy), x0 = (int)floorf(fx);
    const float ty = fy - (float)y0, tx = fx - (float)x0;
    float wy[4], wx[4];
#pragma unroll
    for (int j = 0; j < 4; ++j) {
      wy[j] = cubw(ty - (float)(j - 1));
      wx[j] = cubw(tx - (float)(j - 1));
    }
    float acc = 0.f;
#pragma unroll
    for (int j = 0; j < 4; ++j) {
      int ys = y0 - 1 + j; ys = ys < 0 ? 0 : (ys > SIDE - 1 ? SIDE - 1 : ys);
#pragma unroll
      for (int i = 0; i < 4; ++i) {
        int xs = x0 - 1 + i; xs = xs < 0 ? 0 : (xs > SIDE - 1 ? SIDE - 1 : xs);
        acc += wy[j] * wx[i] * table[(ys * SIDE + xs) * NHEAD + hh];
      }
    }
    out[hh * (TDIM * TDIM) + p] = acc;
  }
}

// ---------------- Kernel 2: Q/K/V projection GEMM ----------------
// Y[r, n] = (sum_k X[r,k] * W[n,k] + bias[n]) * scale
// X: [seq*BSZ, 256] f32 (row = t*BSZ + b), out f16 in [b][h][seq][d].

__global__ void __launch_bounds__(256)
proj_qkv_kernel(const float* __restrict__ X, const float* __restrict__ W,
                const float* __restrict__ bias, _Float16* __restrict__ out,
                int seq, float scale) {
  __shared__ __align__(16) _Float16 As[64 * LDA];
  __shared__ __align__(16) _Float16 Ws[64 * LDA];
  const int tid = threadIdx.x;
  const int r0 = blockIdx.x * 64;
  const int n0 = blockIdx.y * 64;
  const int w  = tid >> 5;
  const int rg = (w & 3) * 16;        // wave's 16-row group
  const int cg = (w >> 2) * 32;       // wave's 32-col group (two 16-wide tiles)
  v8f acc0 = {}, acc1 = {};
  for (int k0 = 0; k0 < E_DIM; k0 += 32) {
    __syncthreads();
    // fp32 -> f16 staging, 128-bit global loads, 64-bit LDS stores
    for (int e = tid * 4; e < 64 * 32; e += 256 * 4) {
      const int rr = e >> 5, kk = e & 31;
      const float4 xa = *(const float4*)&X[(size_t)(r0 + rr) * E_DIM + k0 + kk];
      const float4 xw = *(const float4*)&W[(size_t)(n0 + rr) * E_DIM + k0 + kk];
      v4h ha = {(_Float16)xa.x, (_Float16)xa.y, (_Float16)xa.z, (_Float16)xa.w};
      v4h hw = {(_Float16)xw.x, (_Float16)xw.y, (_Float16)xw.z, (_Float16)xw.w};
      *(v4h*)&As[rr * LDA + kk] = ha;
      *(v4h*)&Ws[rr * LDA + kk] = hw;
    }
    __syncthreads();
    v16h a  = frag_a(As, rg, 0, LDA);
    v16h b0 = frag_bt(Ws, cg, 0, LDA);
    v16h b1 = frag_bt(Ws, cg + 16, 0, LDA);
    acc0 = wmma16(a, b0, acc0);
    acc1 = wmma16(a, b1, acc1);
  }
  const int lane = tid & 31, g = lane >> 4, n = lane & 15;
#pragma unroll
  for (int v = 0; v < 8; ++v) {
    const int row = r0 + rg + v + 8 * g;   // = t*BSZ + b
    const int t = row >> 7, b = row & (BSZ - 1);
#pragma unroll
    for (int sub = 0; sub < 2; ++sub) {
      const int col = n0 + cg + sub * 16 + n;
      const int hh = col >> 5, dd = col & (HDIM - 1);
      const float val = ((sub == 0) ? acc0[v] : acc1[v]) + bias[col];
      out[(((size_t)b * NHEAD + hh) * (size_t)seq + t) * HDIM + dd] =
          (_Float16)(val * scale);
    }
  }
}

// ---------------- Kernel 3: attention core ----------------
// One block = (b,h) pair x 64-query chunk. S = Q K^T (+bias), softmax, O = P V.
// Q/K/V tiles staged with CDNA5 async global->LDS (ASYNCcnt).

__global__ void __launch_bounds__(256)
attn_kernel(const _Float16* __restrict__ qh, const _Float16* __restrict__ kh,
            const _Float16* __restrict__ vh, const float* __restrict__ rpb,
            _Float16* __restrict__ attn_out) {
  __shared__ __align__(16) _Float16 Qs[64 * LDA];
  __shared__ __align__(16) _Float16 Ks[256 * LDA];
  __shared__ __align__(16) _Float16 Vs[256 * LDA];
  __shared__ float Ss[64 * 257];
  const int tid = threadIdx.x;
  const int q0  = blockIdx.x * 64;
  const int bh  = blockIdx.y;                 // b*NHEAD + h
  const int hh  = bh & (NHEAD - 1);
  const size_t qbase = ((size_t)bh * TGT + q0) * HDIM;
  const size_t kbase = (size_t)bh * SRC * HDIM;
  // Q: 64 rows x 64 B  -> 256 16-byte chunks, one per thread
  {
    const int rr = tid >> 2, cc = (tid & 3) * 8;
    async_copy_b128(&Qs[rr * LDA + cc], &qh[qbase + (size_t)rr * HDIM + cc]);
  }
  // K, V: 256 rows x 64 B -> 1024 chunks each, 4 per thread
  for (int c = tid; c < 256 * 4; c += 256) {
    const int rr = c >> 2, cc = (c & 3) * 8;
    async_copy_b128(&Ks[rr * LDA + cc], &kh[kbase + (size_t)rr * HDIM + cc]);
    async_copy_b128(&Vs[rr * LDA + cc], &vh[kbase + (size_t)rr * HDIM + cc]);
  }
  async_wait0();
  __syncthreads();
  // ---- scores: each wave owns 16 rows x 128 cols; K = d = 32 -> 1 WMMA/tile
  {
    const int w  = tid >> 5;
    const int rg = (w & 3) * 16;
    const int c0 = (w >> 2) * 128;
    const int lane = tid & 31, g = lane >> 4, n = lane & 15;
    const v16h a = frag_a(Qs, rg, 0, LDA);
#pragma unroll
    for (int ct = 0; ct < 8; ++ct) {
      const int col = c0 + ct * 16;
      v16h b = frag_bt(Ks, col, 0, LDA);
      v8f acc = {};
      acc = wmma16(a, b, acc);
#pragma unroll
      for (int v = 0; v < 8; ++v)
        Ss[(rg + v + 8 * g) * 257 + col + n] = acc[v];
    }
  }
  __syncthreads();
  // ---- bias + softmax: one thread per query row
  if (tid < 64) {
    const int qg = q0 + tid;
    const int qy = qg / QW, qx = qg % QW;
    float* row = &Ss[tid * 257];
    const float* br = &rpb[hh * (TDIM * TDIM) + (qy + 15) * TDIM + (qx + 15)];
    float mx = -3.0e38f;
    for (int s = 0; s < SRC; ++s) {
      const int ky = s >> 4, kx = s & 15;
      const float val = row[s] + br[-(ky * TDIM + kx)];
      row[s] = val;
      mx = fmaxf(mx, val);
    }
    float sum = 0.f;
    for (int s = 0; s < SRC; ++s) {
      const float ee = __expf(row[s] - mx);
      row[s] = ee;
      sum += ee;
    }
    const float inv = 1.f / sum;
    for (int s = 0; s < SRC; ++s) row[s] *= inv;
  }
  __syncthreads();
  // ---- O = P @ V : 8 tiles of 16x16, K loop over src (8 WMMAs/tile)
  {
    const int w   = tid >> 5;
    const int rg  = (w & 3) * 16;
    const int cgd = (w >> 2) * 16;   // 0 or 16 within d=32
    const int lane = tid & 31, g = lane >> 4, n = lane & 15;
    v8f acc = {};
#pragma unroll
    for (int ks = 0; ks < SRC; ks += 32) {
      v16h a = frag_a_f32(Ss, rg, ks, 257);
      v16h b = frag_bn(Vs, ks, cgd, LDA);
      acc = wmma16(a, b, acc);
    }
    const int b_ = bh >> 3;
#pragma unroll
    for (int v = 0; v < 8; ++v) {
      const int qg = q0 + rg + v + 8 * g;
      const int dd = cgd + n;
      attn_out[((size_t)qg * BSZ + b_) * E_DIM + hh * HDIM + dd] = (_Float16)acc[v];
    }
  }
}

// ---------------- Kernel 4: output projection ----------------
// out[r, n] = sum_k A[r,k] * W[n,k] + bias[n]; A f16 [tgt*BSZ, 256], out f32.
// A tile staged via async global->LDS; W staged through cvt path.

__global__ void __launch_bounds__(256)
proj_out_kernel(const _Float16* __restrict__ A, const float* __restrict__ W,
                const float* __restrict__ bias, float* __restrict__ out) {
  __shared__ __align__(16) _Float16 As[64 * LDA];
  __shared__ __align__(16) _Float16 Ws[64 * LDA];
  const int tid = threadIdx.x;
  const int r0 = blockIdx.x * 64;
  const int n0 = blockIdx.y * 64;
  const int w  = tid >> 5;
  const int rg = (w & 3) * 16;
  const int cg = (w >> 2) * 32;
  v8f acc0 = {}, acc1 = {};
  for (int k0 = 0; k0 < E_DIM; k0 += 32) {
    __syncthreads();
    // A (f16): 64 rows x 64 B -> 256 chunks, async copy
    {
      const int rr = tid >> 2, cc = (tid & 3) * 8;
      async_copy_b128(&As[rr * LDA + cc],
                      &A[(size_t)(r0 + rr) * E_DIM + k0 + cc]);
    }
    // W (f32 -> f16): vectorized cvt path
    for (int e = tid * 4; e < 64 * 32; e += 256 * 4) {
      const int rr = e >> 5, kk = e & 31;
      const float4 xw = *(const float4*)&W[(size_t)(n0 + rr) * E_DIM + k0 + kk];
      v4h hw = {(_Float16)xw.x, (_Float16)xw.y, (_Float16)xw.z, (_Float16)xw.w};
      *(v4h*)&Ws[rr * LDA + kk] = hw;
    }
    async_wait0();
    __syncthreads();
    v16h a  = frag_a(As, rg, 0, LDA);
    v16h b0 = frag_bt(Ws, cg, 0, LDA);
    v16h b1 = frag_bt(Ws, cg + 16, 0, LDA);
    acc0 = wmma16(a, b0, acc0);
    acc1 = wmma16(a, b1, acc1);
  }
  const int lane = tid & 31, g = lane >> 4, n = lane & 15;
#pragma unroll
  for (int v = 0; v < 8; ++v) {
    const int row = r0 + rg + v + 8 * g;
#pragma unroll
    for (int sub = 0; sub < 2; ++sub) {
      const int col = n0 + cg + sub * 16 + n;
      out[(size_t)row * E_DIM + col] =
          ((sub == 0) ? acc0[v] : acc1[v]) + bias[col];
    }
  }
}

// ---------------- Host launcher ----------------

extern "C" void kernel_launch(void* const* d_in, const int* in_sizes, int n_in,
                              void* d_out, int out_size, void* d_ws, size_t ws_size,
                              hipStream_t stream) {
  (void)in_sizes; (void)n_in; (void)out_size; (void)ws_size;
  const float* query = (const float*)d_in[0];
  const float* key   = (const float*)d_in[1];
  const float* value = (const float*)d_in[2];
  const float* wq = (const float*)d_in[3];
  const float* bq = (const float*)d_in[4];
  const float* wk = (const float*)d_in[5];
  const float* bk = (const float*)d_in[6];
  const float* wv = (const float*)d_in[7];
  const float* bv = (const float*)d_in[8];
  const float* wo = (const float*)d_in[9];
  const float* bo = (const float*)d_in[10];
  const float* rpb_table = (const float*)d_in[11];
  float* out = (float*)d_out;

  // Scratch carve-out.
  char* ws = (char*)d_ws;
  size_t off = 0;
  auto carve = [&](size_t bytes) {
    void* p = ws + off;
    off += (bytes + 255) & ~(size_t)255;
    return p;
  };
  _Float16* qh   = (_Float16*)carve((size_t)BSZ * NHEAD * TGT * HDIM * 2);
  _Float16* kh   = (_Float16*)carve((size_t)BSZ * NHEAD * SRC * HDIM * 2);
  _Float16* vh   = (_Float16*)carve((size_t)BSZ * NHEAD * SRC * HDIM * 2);
  _Float16* attn = (_Float16*)carve((size_t)TGT * BSZ * E_DIM * 2);
  float*    rpbr = (float*)carve((size_t)NHEAD * TDIM * TDIM * 4);

  const float qscale = 0.1767766953f;  // 1/sqrt(d), d = 32

  rpe_resize_kernel<<<dim3(NHEAD), dim3(256), 0, stream>>>(rpb_table, rpbr);

  proj_qkv_kernel<<<dim3((TGT * BSZ) / 64, E_DIM / 64), 256, 0, stream>>>(
      query, wq, bq, qh, TGT, qscale);
  proj_qkv_kernel<<<dim3((SRC * BSZ) / 64, E_DIM / 64), 256, 0, stream>>>(
      key, wk, bk, kh, SRC, 1.0f);
  proj_qkv_kernel<<<dim3((SRC * BSZ) / 64, E_DIM / 64), 256, 0, stream>>>(
      value, wv, bv, vh, SRC, 1.0f);

  attn_kernel<<<dim3(TGT / 64, BSZ * NHEAD), 256, 0, stream>>>(
      qh, kh, vh, rpbr, attn);

  proj_out_kernel<<<dim3((TGT * BSZ) / 64, E_DIM / 64), 256, 0, stream>>>(
      attn, wo, bo, out);
}
